// GNNAggregationModule_4947802324998
// MI455X (gfx1250) — matrix operand order repaired and checked by reading, not compiled, not run on hardware
//
#include <hip/hip_runtime.h>
#include <hip/hip_bf16.h>

typedef __bf16 bf16;
typedef __attribute__((ext_vector_type(16))) __bf16 bf16x16;
typedef __attribute__((ext_vector_type(8)))  __bf16 bf16x8;
typedef __attribute__((ext_vector_type(8)))  float  f32x8;
typedef __attribute__((ext_vector_type(4)))  unsigned int u32x4;

#define KN     32
#define DIM    256
#define LAYERS 3
#define NPB    2      // batch nodes per workgroup (W-fragment reuse x2)

// LDS row pitches (elements), padded so 16-lane b128 accesses spread across banks
#define H_PITCH  (DIM + 8)   // 264 bf16 = 528 B
#define XT_PITCH (KN + 8)    // 40 bf16 = 80 B
#define A_PITCH  (KN + 8)

__device__ __forceinline__ float fast_tanh(float x) {
#if __has_builtin(__builtin_amdgcn_tanhf)
  return __builtin_amdgcn_tanhf(x);   // gfx1250 v_tanh_f32 (trans op)
#else
  return tanhf(x);
#endif
}

// 16-element bf16 fragment from two 16B chunks: [0..7] at base, [8..15] at base+ofs2
__device__ __forceinline__ bf16x16 load_frag2x16(const bf16* base, int ofs2) {
  u32x4 lo = *reinterpret_cast<const u32x4*>(base);
  u32x4 hi = *reinterpret_cast<const u32x4*>(base + ofs2);
  bf16x8 a = __builtin_bit_cast(bf16x8, lo);
  bf16x8 b = __builtin_bit_cast(bf16x8, hi);
  bf16x16 r;
#pragma unroll
  for (int i = 0; i < 8; ++i) { r[i] = a[i]; r[i + 8] = b[i]; }
  return r;
}

// B fragment from a contiguous fp32 row of W (16 floats/lane, convert in-flight)
__device__ __forceinline__ bf16x16 load_bfrag_f32(const float* wrow) {
  const float4* p = reinterpret_cast<const float4*>(wrow);
  bf16x16 r;
#pragma unroll
  for (int i = 0; i < 4; ++i) {
    float4 v = p[i];
    r[4 * i + 0] = (bf16)v.x; r[4 * i + 1] = (bf16)v.y;
    r[4 * i + 2] = (bf16)v.z; r[4 * i + 3] = (bf16)v.w;
  }
  return r;
}

// ---- pre-pass: W fp32 -> bf16 in workspace (run once per launch; 384 KB) ----
__global__ void convert_w_bf16(const float* __restrict__ W, bf16* __restrict__ Wb, int n) {
  int i = (blockIdx.x * 256 + threadIdx.x) * 4;
  if (i + 3 < n) {
    float4 v = *reinterpret_cast<const float4*>(W + i);
    bf16* d = Wb + i;
    d[0] = (bf16)v.x; d[1] = (bf16)v.y; d[2] = (bf16)v.z; d[3] = (bf16)v.w;
  }
}

template <bool WBF16>
__global__ __launch_bounds__(256)
void gnn_agg_kernel(const int* __restrict__ node_indexes,
                    const int* __restrict__ neighbors,
                    const float* __restrict__ As,
                    const float* __restrict__ emb,
                    const float* __restrict__ Wf,
                    const bf16* __restrict__ Wb,
                    const float* __restrict__ bias,
                    float* __restrict__ out,
                    int nbatch) {
  __shared__ bf16 sh_h[NPB][KN * H_PITCH];     // h, row-major
  __shared__ bf16 sh_xt[NPB][DIM * XT_PITCH];  // X^T, row-major
  __shared__ bf16 sh_a[NPB][KN * A_PITCH];     // sub_A bf16

  const int tid  = threadIdx.x;
  const int lane = tid & 31;
  const int wave = tid >> 5;              // 0..7
  const int b0   = blockIdx.x * NPB;

  // ---- gather sub_A and h for both nodes
#pragma unroll
  for (int j = 0; j < NPB; ++j) {
    const int bi   = min(b0 + j, nbatch - 1);
    const int node = node_indexes[bi];
    {
      const float* Ap = As + (size_t)node * KN * KN;
      int i = tid * 4;
      float4 v = *reinterpret_cast<const float4*>(Ap + i);
      bf16* d = &sh_a[j][(i >> 5) * A_PITCH + (i & 31)];
      d[0] = (bf16)v.x; d[1] = (bf16)v.y; d[2] = (bf16)v.z; d[3] = (bf16)v.w;
    }
    {
      int row = tid >> 3;
      int col = (tid & 7) * 32;
      int nb  = neighbors[node * KN + row];
      const float* src = emb + (size_t)nb * DIM + col;
      bf16* dst = &sh_h[j][row * H_PITCH + col];
#pragma unroll
      for (int c = 0; c < 32; c += 4) {
        float4 v = *reinterpret_cast<const float4*>(src + c);
        dst[c + 0] = (bf16)v.x; dst[c + 1] = (bf16)v.y;
        dst[c + 2] = (bf16)v.z; dst[c + 3] = (bf16)v.w;
      }
    }
  }
  __syncthreads();

  const int lm   = lane & 15;
  const int aofs = (lane & 16);       // A-frag: lo lanes K 0..15, hi 16..31
  const int bofs = (lane & 16);       // B-frag: lo lanes K 0..15, hi 16..31

  // sub_A A-fragments are layer-invariant: hoist into registers
  bf16x16 aS[NPB][2];
#pragma unroll
  for (int j = 0; j < NPB; ++j) {
    aS[j][0] = load_frag2x16(&sh_a[j][(0  + lm) * A_PITCH + aofs], 8);
    aS[j][1] = load_frag2x16(&sh_a[j][(16 + lm) * A_PITCH + aofs], 8);
  }

  const int e0base = wave * 32;       // this wave owns output columns [e0base, e0base+32)
  float colsum[NPB][2] = {};

  for (int l = 0; l < LAYERS; ++l) {
    const float* bl = bias + l * DIM;

    // ---------- stage 1: X = h @ W^T + b  -> X^T (bf16) in LDS ----------
#pragma unroll
    for (int ct = 0; ct < 2; ++ct) {
      const int e0 = e0base + ct * 16;
      const float bv = bl[e0 + lm];
      f32x8 acc[NPB][2];
#pragma unroll
      for (int j = 0; j < NPB; ++j)
#pragma unroll
        for (int r = 0; r < 2; ++r)
#pragma unroll
          for (int i = 0; i < 8; ++i) acc[j][r][i] = bv;

#pragma unroll
      for (int d0 = 0; d0 < DIM; d0 += 32) {
        bf16x16 bt;  // one W fragment feeds NPB*2 = 4 WMMAs
        if constexpr (WBF16) {
          bt = load_frag2x16(Wb + (size_t)l * DIM * DIM + (size_t)(e0 + lm) * DIM + bofs + d0, 8);
        } else {
          bt = load_bfrag_f32(Wf + (size_t)l * DIM * DIM + (size_t)(e0 + lm) * DIM + bofs + d0);
        }
#pragma unroll
        for (int j = 0; j < NPB; ++j) {
          bf16x16 a0 = load_frag2x16(&sh_h[j][(0  + lm) * H_PITCH + d0 + aofs], 8);
          bf16x16 a1 = load_frag2x16(&sh_h[j][(16 + lm) * H_PITCH + d0 + aofs], 8);
          acc[j][0] = __builtin_amdgcn_wmma_f32_16x16x32_bf16(false, a0, false, bt,
                                                              (short)0, acc[j][0], false, false);
          acc[j][1] = __builtin_amdgcn_wmma_f32_16x16x32_bf16(false, a1, false, bt,
                                                              (short)0, acc[j][1], false, false);
        }
      }
      // store D tiles transposed: lane writes 8 contiguous bf16 of X^T row (e0+lm)
      const int khalf = (lane & 16) >> 1;  // 0 or 8
#pragma unroll
      for (int j = 0; j < NPB; ++j) {
        bf16* xr = &sh_xt[j][(e0 + lm) * XT_PITCH];
        bf16x8 p0, p1;
#pragma unroll
        for (int i = 0; i < 8; ++i) { p0[i] = (bf16)acc[j][0][i]; p1[i] = (bf16)acc[j][1][i]; }
        *reinterpret_cast<u32x4*>(xr + 0  + khalf) = __builtin_bit_cast(u32x4, p0);
        *reinterpret_cast<u32x4*>(xr + 16 + khalf) = __builtin_bit_cast(u32x4, p1);
      }
    }
    if (l + 1 < LAYERS) {  // warm caches for next layer's W rows this wave will read
      if constexpr (WBF16)
        __builtin_prefetch(Wb + (size_t)(l + 1) * DIM * DIM + (size_t)(e0base + lm) * DIM, 0, 0);
      else
        __builtin_prefetch(Wf + (size_t)(l + 1) * DIM * DIM + (size_t)(e0base + lm) * DIM, 0, 0);
    }
    __syncthreads();

    // ---------- stage 2: h' = tanh(sub_A @ X), K = 32 in one WMMA ----------
#pragma unroll
    for (int ct = 0; ct < 2; ++ct) {
      const int e0 = e0base + ct * 16;
#pragma unroll
      for (int j = 0; j < NPB; ++j) {
        // B fragment = column (e0+lm) of X = contiguous row of X^T
        bf16x16 bt = load_frag2x16(&sh_xt[j][(e0 + lm) * XT_PITCH + bofs], 8);
        f32x8 z = {};
        f32x8 h0 = __builtin_amdgcn_wmma_f32_16x16x32_bf16(false, aS[j][0], false, bt,
                                                           (short)0, z, false, false);
        f32x8 h1 = __builtin_amdgcn_wmma_f32_16x16x32_bf16(false, aS[j][1], false, bt,
                                                           (short)0, z, false, false);
#pragma unroll
        for (int i = 0; i < 8; ++i) { h0[i] = fast_tanh(h0[i]); h1[i] = fast_tanh(h1[i]); }

        if (l < LAYERS - 1) {
          const int krow = (lane & 16) >> 1;  // 0 or 8
#pragma unroll
          for (int i = 0; i < 8; ++i) {
            sh_h[j][(krow + i)      * H_PITCH + e0 + lm] = (bf16)h0[i];
            sh_h[j][(16 + krow + i) * H_PITCH + e0 + lm] = (bf16)h1[i];
          }
        } else {
          float s = 0.f;
#pragma unroll
          for (int i = 0; i < 8; ++i) s += h0[i] + h1[i];
          colsum[j][ct] = s;
        }
      }
    }
    __syncthreads();
  }

  // fold the two k-halves held by lane pairs (n, n+16), then sum-pool store
#pragma unroll
  for (int j = 0; j < NPB; ++j) {
    colsum[j][0] += __shfl_xor(colsum[j][0], 16, 32);
    colsum[j][1] += __shfl_xor(colsum[j][1], 16, 32);
    if (lane < 16 && (b0 + j) < nbatch) {
      float* orow = out + (size_t)(b0 + j) * DIM + e0base;
      orow[lane]      = colsum[j][0];
      orow[16 + lane] = colsum[j][1];
    }
  }
}

extern "C" void kernel_launch(void* const* d_in, const int* in_sizes, int n_in,
                              void* d_out, int out_size, void* d_ws, size_t ws_size,
                              hipStream_t stream) {
  const int*   node_indexes = (const int*)d_in[0];
  const int*   neighbors    = (const int*)d_in[1];
  const float* As           = (const float*)d_in[2];
  const float* emb          = (const float*)d_in[3];
  const float* W            = (const float*)d_in[4];
  const float* bias         = (const float*)d_in[5];
  float*       out          = (float*)d_out;

  const int B     = in_sizes[0];
  const int nwe   = LAYERS * DIM * DIM;            // W element count
  const int nblk  = (B + NPB - 1) / NPB;

  if (ws_size >= (size_t)nwe * sizeof(bf16)) {
    bf16* Wb = (bf16*)d_ws;
    convert_w_bf16<<<dim3((nwe / 4 + 255) / 256), dim3(256), 0, stream>>>(W, Wb, nwe);
    gnn_agg_kernel<true><<<dim3(nblk), dim3(256), 0, stream>>>(
        node_indexes, neighbors, As, emb, W, Wb, bias, out, B);
  } else {
    gnn_agg_kernel<false><<<dim3(nblk), dim3(256), 0, stream>>>(
        node_indexes, neighbors, As, emb, W, (const bf16*)nullptr, bias, out, B);
  }
}
